// MCAAttention_82669530514245
// MI455X (gfx1250) — compile-verified, matrix operand order checked by reading, not run on hardware
//
#include <hip/hip_runtime.h>
#include <hip/hip_bf16.h>

// ---------------------------------------------------------------------------
// MCAAttention for MI455X (gfx1250, wave32, WMMA + async-to-LDS staging).
//  Shapes fixed by the reference: B=2, N=2048, C=1024, H=16, d=64.
//  Strategy:
//   1) LayerNorm -> x1 (f32, ws)
//   2) fused 3-scale depthwise convs (K=7/11/21), both branches -> conv1/2 (bf16)
//   3) WMMA bf16 GEMM: out1 = conv1 @ W^T + b, out2 = conv2 @ W^T + b (f32)
//   4) per-head L2 norms -> o1n/o2n (bf16) + transposed V tensors v1t/v2t (bf16)
//   5) WMMA GEMM (batched over B*H): S = o2n_h @ o1n_h^T  (attn1 slot of d_out,
//      pre-softmax).  S2 = S^T, so one GEMM serves both attention maps.
//   6) column softmax of S -> attn2 (bf16, ws, via LDS transpose)
//   7) row softmax of S in place -> attn1 (f32, final output)
//   8) WMMA GEMM: m  = attn1 @ v1 + q1   (A loaded f32, converted to bf16)
//      WMMA GEMM: m += attn2 @ v2 + q2
//   9) WMMA GEMM: out = m @ W^T + 2*proj_b + x
//  bf16 GEMM tiles are staged with GLOBAL_LOAD_ASYNC_TO_LDS_B128 (ASYNCcnt),
//  bypassing VGPRs; f32 attn tiles convert through VGPRs.
//  Workspace requirement ~373 MB (dominated by attn2 bf16, 256 MB).
// ---------------------------------------------------------------------------

typedef __attribute__((ext_vector_type(16))) __bf16 v16bf;
typedef __attribute__((ext_vector_type(8)))  float  v8f;

#define BB   2
#define NN   2048
#define CC   1024
#define HH   16
#define DD   64

// ---- CDNA5 async memory->LDS copy (CDNA5 ISA) -------------
// VDST = per-lane LDS byte offset (low 32 bits of the generic pointer: the
// LDS aperture maps addr[31:0] to the wave's LDS offset), VADDR = 64-bit
// global address.  Tracked by ASYNCcnt; drained with s_wait_asynccnt.
__device__ __forceinline__ void async_load_b128(uint32_t lds_off, const void* gptr) {
  asm volatile("global_load_async_to_lds_b128 %0, %1, off"
               :: "v"(lds_off), "v"(gptr)
               : "memory");
}
__device__ __forceinline__ void wait_asynccnt0() {
  asm volatile("s_wait_asynccnt 0x0" ::: "memory");
}
__device__ __forceinline__ uint32_t lds_off32(const void* p) {
  return (uint32_t)(uintptr_t)p;
}

// ------------------------------ LayerNorm ----------------------------------
__global__ __launch_bounds__(256) void ln_kernel(const float* __restrict__ x,
                                                 const float* __restrict__ g,
                                                 const float* __restrict__ bta,
                                                 float* __restrict__ x1) {
  __shared__ float s1[256], s2[256];
  const size_t row = blockIdx.x;               // b*N + n
  const int t = threadIdx.x;
  const float* xr = x + row * CC;
  float v[4], lsum = 0.f, lsq = 0.f;
#pragma unroll
  for (int i = 0; i < 4; ++i) {
    v[i] = xr[t + 256 * i];
    lsum += v[i];
    lsq  += v[i] * v[i];
  }
  s1[t] = lsum; s2[t] = lsq;
  __syncthreads();
  for (int off = 128; off > 0; off >>= 1) {
    if (t < off) { s1[t] += s1[t + off]; s2[t] += s2[t + off]; }
    __syncthreads();
  }
  const float mu  = s1[0] * (1.f / CC);
  const float var = s2[0] * (1.f / CC) - mu * mu;
  const float rs  = rsqrtf(var + 1e-5f);
  float* orow = x1 + row * CC;
#pragma unroll
  for (int i = 0; i < 4; ++i) {
    const int c = t + 256 * i;
    orow[c] = (v[i] - mu) * rs * g[c] + bta[c];
  }
}

// ---------------------- f32 -> bf16 weight conversion ----------------------
__global__ __launch_bounds__(256) void wcvt_kernel(const float* __restrict__ w,
                                                   __bf16* __restrict__ wb, int n) {
  const int i = blockIdx.x * 256 + threadIdx.x;
  if (i < n) wb[i] = (__bf16)w[i];
}

// ---------------- fused multi-scale depthwise conv (both branches) ---------
__global__ __launch_bounds__(256) void dwconv_kernel(
    const float* __restrict__ x1,
    const float* __restrict__ w7a,  const float* __restrict__ b7a,
    const float* __restrict__ w7b,  const float* __restrict__ b7b,
    const float* __restrict__ w11a, const float* __restrict__ b11a,
    const float* __restrict__ w11b, const float* __restrict__ b11b,
    const float* __restrict__ w21a, const float* __restrict__ b21a,
    const float* __restrict__ w21b, const float* __restrict__ b21b,
    __bf16* __restrict__ conv1, __bf16* __restrict__ conv2) {
  __shared__ float sm[84 * 64];                // 64 rows + 2*10 halo, 64 channels
  const int t  = threadIdx.x;
  const int c0 = blockIdx.x * 64;
  const int n0 = blockIdx.y * 64;
  const int b  = blockIdx.z;
  const float* xb = x1 + (size_t)b * NN * CC;
  for (int idx = t; idx < 84 * 64; idx += 256) {
    const int r = idx >> 6, cc = idx & 63;
    const int gn = n0 - 10 + r;
    sm[idx] = (gn >= 0 && gn < NN) ? xb[(size_t)gn * CC + c0 + cc] : 0.f;
  }
  __syncthreads();
  const int cl = t & 63, nl = t >> 6;          // 64 channels x 4 row-slots
  const int gc = c0 + cl;
  float acc1[16], acc2[16];
  const float bias1 = b7a[gc] + b11a[gc] + b21a[gc];
  const float bias2 = b7b[gc] + b11b[gc] + b21b[gc];
#pragma unroll
  for (int i = 0; i < 16; ++i) { acc1[i] = bias1; acc2[i] = bias2; }
#pragma unroll
  for (int k = 0; k < 7; ++k) {                // pad 3 -> local offset +7
    const float wa = w7a[gc * 7 + k], wb = w7b[gc * 7 + k];
#pragma unroll
    for (int i = 0; i < 16; ++i) {
      const float xv = sm[(nl + 4 * i + k + 7) * 64 + cl];
      acc1[i] += wa * xv; acc2[i] += wb * xv;
    }
  }
#pragma unroll
  for (int k = 0; k < 11; ++k) {               // pad 5 -> +5
    const float wa = w11a[gc * 11 + k], wb = w11b[gc * 11 + k];
#pragma unroll
    for (int i = 0; i < 16; ++i) {
      const float xv = sm[(nl + 4 * i + k + 5) * 64 + cl];
      acc1[i] += wa * xv; acc2[i] += wb * xv;
    }
  }
#pragma unroll
  for (int k = 0; k < 21; ++k) {               // pad 10 -> +0
    const float wa = w21a[gc * 21 + k], wb = w21b[gc * 21 + k];
#pragma unroll
    for (int i = 0; i < 16; ++i) {
      const float xv = sm[(nl + 4 * i + k) * 64 + cl];
      acc1[i] += wa * xv; acc2[i] += wb * xv;
    }
  }
#pragma unroll
  for (int i = 0; i < 16; ++i) {
    const size_t o = ((size_t)b * NN + n0 + nl + 4 * i) * CC + gc;
    conv1[o] = (__bf16)acc1[i];
    conv2[o] = (__bf16)acc2[i];
  }
}

// --------------------------- generic WMMA TN GEMM --------------------------
// D[M,Nout] = A[M,K] @ B[Nout,K]^T  (+ bias*scale + bf16 add + f32 add + accum)
// 8 waves/block; each wave computes a 32x32 tile via 2x2 v_wmma_f32_16x16x32_bf16.
// bf16 operands are staged via global_load_async_to_lds_b128 (ASYNCcnt path).
template <int WM, int WN, bool A_IS_F32>
__global__ __launch_bounds__(256) void gemm_tn_wmma(
    const void* __restrict__ Aptr, const __bf16* __restrict__ Bptr,
    float* __restrict__ Dptr,
    int M, int Nout, int K, int lda, int ldb, int ldd,
    long sA_o, long sA_i, long sB_o, long sB_i, long sD_o, long sD_i, int Hb,
    const float* __restrict__ bias, float bias_scale,
    const float* __restrict__ add_f32, const __bf16* __restrict__ addq,
    int accum) {
  constexpr int BM = WM * 32, BN = WN * 32, BK = 32, LSTR = BK + 8; // pad -> no bank conflicts
  __shared__ __bf16 As[BM * LSTR];
  __shared__ __bf16 Bs[BN * LSTR];
  const int tid = threadIdx.x;
  const int z  = blockIdx.z;
  const int zo = z / Hb, zi = z % Hb;
  const long aBase = (long)zo * sA_o + (long)zi * sA_i;
  const long bBase = (long)zo * sB_o + (long)zi * sB_i;
  const long dBase = (long)zo * sD_o + (long)zi * sD_i;
  const int m0 = blockIdx.y * BM;
  const int n0 = blockIdx.x * BN;

  const __bf16* Abf = (const __bf16*)Aptr + aBase;
  const float*  Af  = (const float*)Aptr + aBase;
  const __bf16* Bb  = Bptr + bBase;

  const uint32_t asBase = lds_off32(As);
  const uint32_t bsBase = lds_off32(Bs);

  const int lane = tid & 31;
  const int w    = tid >> 5;
  const int tm   = (w / WN) * 32;
  const int tn   = (w % WN) * 32;
  const int lr   = lane & 15;
  const int kb   = (lane >> 4) * 8;            // K-half selector per ISA A layout

  v8f acc00 = {}, acc01 = {}, acc10 = {}, acc11 = {};

  for (int k0 = 0; k0 < K; k0 += BK) {
    // ---- stage A tile (async for bf16; VGPR convert path for f32) ----
#pragma unroll
    for (int it = 0; it < BM / 64; ++it) {
      const int e = (tid + it * 256) * 8;
      const int r = e >> 5, c = e & 31;
      if constexpr (A_IS_F32) {
        const float* src = Af + (size_t)(m0 + r) * lda + (k0 + c);
        const float4 f0 = *(const float4*)src;
        const float4 f1 = *(const float4*)(src + 4);
        __bf16* d = &As[r * LSTR + c];
        d[0] = (__bf16)f0.x; d[1] = (__bf16)f0.y; d[2] = (__bf16)f0.z; d[3] = (__bf16)f0.w;
        d[4] = (__bf16)f1.x; d[5] = (__bf16)f1.y; d[6] = (__bf16)f1.z; d[7] = (__bf16)f1.w;
      } else {
        async_load_b128(asBase + (uint32_t)(r * LSTR + c) * 2,
                        Abf + (size_t)(m0 + r) * lda + (k0 + c));
      }
    }
    // ---- stage B tile (always bf16 -> async direct-to-LDS) ----
#pragma unroll
    for (int it = 0; it < BN / 64; ++it) {
      const int e = (tid + it * 256) * 8;
      const int r = e >> 5, c = e & 31;
      async_load_b128(bsBase + (uint32_t)(r * LSTR + c) * 2,
                      Bb + (size_t)(n0 + r) * ldb + (k0 + c));
    }
    // prefetch next B tile row for this thread (global_prefetch_b8)
    if (k0 + BK < K)
      __builtin_prefetch((const void*)(Bb + (size_t)(n0 + (tid & (BN - 1))) * ldb + k0 + BK), 0, 1);
    wait_asynccnt0();                          // drain this wave's async copies
    __syncthreads();

    union Frag { uint4 u[2]; v16bf v; };
    Frag a0, a1, b0, b1;
    a0.u[0] = *(const uint4*)&As[(tm + lr) * LSTR + kb];
    a0.u[1] = *(const uint4*)&As[(tm + lr) * LSTR + 16 + kb];
    a1.u[0] = *(const uint4*)&As[(tm + 16 + lr) * LSTR + kb];
    a1.u[1] = *(const uint4*)&As[(tm + 16 + lr) * LSTR + 16 + kb];
    b0.u[0] = *(const uint4*)&Bs[(tn + lr) * LSTR + kb];
    b0.u[1] = *(const uint4*)&Bs[(tn + lr) * LSTR + 16 + kb];
    b1.u[0] = *(const uint4*)&Bs[(tn + 16 + lr) * LSTR + kb];
    b1.u[1] = *(const uint4*)&Bs[(tn + 16 + lr) * LSTR + 16 + kb];

    acc00 = __builtin_amdgcn_wmma_f32_16x16x32_bf16(false, a0.v, false, b0.v, (short)0, acc00, false, false);
    acc01 = __builtin_amdgcn_wmma_f32_16x16x32_bf16(false, a0.v, false, b1.v, (short)0, acc01, false, false);
    acc10 = __builtin_amdgcn_wmma_f32_16x16x32_bf16(false, a1.v, false, b0.v, (short)0, acc10, false, false);
    acc11 = __builtin_amdgcn_wmma_f32_16x16x32_bf16(false, a1.v, false, b1.v, (short)0, acc11, false, false);
    __syncthreads();
  }

  // ---- epilogue: C/D layout -> row = (lane>>4)*8 + vgpr, col = lane&15 ----
  float* D = Dptr + dBase;
  const int rowadd = (lane >> 4) * 8;
  auto store_tile = [&](const v8f& a, int i, int j) {
#pragma unroll
    for (int r = 0; r < 8; ++r) {
      const long grow = m0 + tm + 16 * i + rowadd + r;
      const long gcol = n0 + tn + 16 * j + lr;
      const size_t idx = (size_t)grow * ldd + gcol;
      float v = a[r];
      if (bias)    v += bias_scale * bias[gcol];
      if (addq)    v += (float)addq[(size_t)dBase + idx];
      if (add_f32) v += add_f32[(size_t)dBase + idx];
      if (accum)   v += D[idx];
      D[idx] = v;
    }
  };
  store_tile(acc00, 0, 0); store_tile(acc01, 0, 1);
  store_tile(acc10, 1, 0); store_tile(acc11, 1, 1);
}

// ------------- per-head L2 norms + bf16 copies + transposed V --------------
__global__ __launch_bounds__(256) void normsplit_kernel(
    const float* __restrict__ out1, const float* __restrict__ out2,
    __bf16* __restrict__ o1n, __bf16* __restrict__ o2n,
    __bf16* __restrict__ v1t, __bf16* __restrict__ v2t) {
  const size_t row = blockIdx.x;               // b*N + n
  const int b = (int)(row >> 11), n = (int)(row & 2047);
  const int t = threadIdx.x;
  const int c = t * 4;                         // 4 consecutive channels, inside one head
  const int h = t >> 4;                        // 16 threads per head
  const float* p1 = out1 + row * CC + c;
  const float* p2 = out2 + row * CC + c;
  float a[4], q[4], ss1 = 0.f, ss2 = 0.f;
#pragma unroll
  for (int j = 0; j < 4; ++j) {
    a[j] = p1[j]; ss1 += a[j] * a[j];
    q[j] = p2[j]; ss2 += q[j] * q[j];
  }
#pragma unroll
  for (int off = 1; off < 16; off <<= 1) {     // reduce within 16-lane head group
    ss1 += __shfl_xor(ss1, off, 32);
    ss2 += __shfl_xor(ss2, off, 32);
  }
  const float rn1 = 1.f / fmaxf(sqrtf(ss1), 1e-12f);
  const float rn2 = 1.f / fmaxf(sqrtf(ss2), 1e-12f);
  const size_t ob = row * CC + c;
  const int cl = c & 63;
  const size_t vb = (size_t)(b * HH + h) * DD;
#pragma unroll
  for (int j = 0; j < 4; ++j) {
    o1n[ob + j] = (__bf16)(a[j] * rn1);
    o2n[ob + j] = (__bf16)(q[j] * rn2);
    v1t[(vb + cl + j) * NN + n] = (__bf16)a[j];
    v2t[(vb + cl + j) * NN + n] = (__bf16)q[j];
  }
}

// ------------------- column stats of S (for attn2 = softmax(S^T)) ----------
__global__ __launch_bounds__(256) void colstat_kernel(const float* __restrict__ S,
                                                      float* __restrict__ cmax,
                                                      float* __restrict__ csum) {
  const int bh = blockIdx.y;
  const int c  = blockIdx.x * 256 + threadIdx.x;
  const float* Sb = S + (size_t)bh * NN * NN;
  float mx = -3.402823e38f;
  for (int j = 0; j < NN; ++j) mx = fmaxf(mx, Sb[(size_t)j * NN + c]);
  float s = 0.f;
  for (int j = 0; j < NN; ++j) s += expf(Sb[(size_t)j * NN + c] - mx);
  cmax[bh * NN + c] = mx;
  csum[bh * NN + c] = s;
}

// -------- attn2[n][m] = exp(S[m][n]-cmax[n])/csum[n]  via LDS transpose ----
__global__ __launch_bounds__(256) void attn2_kernel(const float* __restrict__ S,
                                                    const float* __restrict__ cmax,
                                                    const float* __restrict__ csum,
                                                    __bf16* __restrict__ attn2) {
  __shared__ __bf16 tile[64 * 68];
  const int bh = blockIdx.z;
  const int j0 = blockIdx.x * 64, c0 = blockIdx.y * 64;
  const float* Sb = S + (size_t)bh * NN * NN;
  const int t = threadIdx.x;
#pragma unroll
  for (int i = 0; i < 16; ++i) {
    const int idx = t + 256 * i;
    const int jl = idx >> 6, cl = idx & 63;
    const float s = Sb[(size_t)(j0 + jl) * NN + (c0 + cl)];
    const float e = expf(s - cmax[bh * NN + c0 + cl]) / csum[bh * NN + c0 + cl];
    tile[cl * 68 + jl] = (__bf16)e;
  }
  __syncthreads();
  __bf16* Ab = attn2 + (size_t)bh * NN * NN;
#pragma unroll
  for (int i = 0; i < 16; ++i) {
    const int idx = t + 256 * i;
    const int cl = idx >> 6, jl = idx & 63;
    Ab[(size_t)(c0 + cl) * NN + (j0 + jl)] = tile[cl * 68 + jl];
  }
}

// --------------------------- row softmax (attn1, in place) -----------------
__global__ __launch_bounds__(256) void rowsoftmax_kernel(float* __restrict__ S) {
  __shared__ float red[256];
  const size_t row = blockIdx.x;
  float* r = S + row * NN;
  const int t = threadIdx.x;
  float v[8], mx = -3.402823e38f;
#pragma unroll
  for (int i = 0; i < 8; ++i) { v[i] = r[t + 256 * i]; mx = fmaxf(mx, v[i]); }
  red[t] = mx; __syncthreads();
  for (int off = 128; off > 0; off >>= 1) {
    if (t < off) red[t] = fmaxf(red[t], red[t + off]);
    __syncthreads();
  }
  mx = red[0]; __syncthreads();
  float s = 0.f;
#pragma unroll
  for (int i = 0; i < 8; ++i) { v[i] = expf(v[i] - mx); s += v[i]; }
  red[t] = s; __syncthreads();
  for (int off = 128; off > 0; off >>= 1) {
    if (t < off) red[t] += red[t + off];
    __syncthreads();
  }
  const float inv = 1.f / red[0];
#pragma unroll
  for (int i = 0; i < 8; ++i) r[t + 256 * i] = v[i] * inv;
}

// ------------------------------- launcher ----------------------------------
extern "C" void kernel_launch(void* const* d_in, const int* in_sizes, int n_in,
                              void* d_out, int out_size, void* d_ws, size_t ws_size,
                              hipStream_t stream) {
  const float* x      = (const float*)d_in[0];
  const float* ln_g   = (const float*)d_in[1];
  const float* ln_b   = (const float*)d_in[2];
  const float* proj_w = (const float*)d_in[3];
  const float* proj_b = (const float*)d_in[4];
  const float* w0_1 = (const float*)d_in[5];  const float* b0_1 = (const float*)d_in[6];
  const float* w0_2 = (const float*)d_in[7];  const float* b0_2 = (const float*)d_in[8];
  const float* w1_1 = (const float*)d_in[9];  const float* b1_1 = (const float*)d_in[10];
  const float* w1_2 = (const float*)d_in[11]; const float* b1_2 = (const float*)d_in[12];
  const float* w2_1 = (const float*)d_in[13]; const float* b2_1 = (const float*)d_in[14];
  const float* w2_2 = (const float*)d_in[15]; const float* b2_2 = (const float*)d_in[16];

  float* out = (float*)d_out;
  float* S   = out + (size_t)BB * NN * CC;     // attn1 slot, used as logit scratch

  const size_t BNC = (size_t)BB * NN * CC;
  char* ws = (char*)d_ws;
  size_t off = 0;
  auto alloc = [&](size_t bytes) -> void* {
    void* p = ws + off;
    off += (bytes + 255) & ~(size_t)255;
    return p;
  };
  float*  x1    = (float*) alloc(BNC * 4);
  __bf16* conv1 = (__bf16*)alloc(BNC * 2);
  __bf16* conv2 = (__bf16*)alloc(BNC * 2);
  __bf16* Wbf   = (__bf16*)alloc((size_t)CC * CC * 2);
  float*  out1  = (float*) alloc(BNC * 4);
  float*  out2  = (float*) alloc(BNC * 4);
  __bf16* o1n   = (__bf16*)alloc(BNC * 2);
  __bf16* o2n   = (__bf16*)alloc(BNC * 2);
  __bf16* v1t   = (__bf16*)alloc(BNC * 2);
  __bf16* v2t   = (__bf16*)alloc(BNC * 2);
  float*  mbuf  = (float*) alloc(BNC * 4);
  float*  cmax  = (float*) alloc((size_t)BB * HH * NN * 4);
  float*  csum  = (float*) alloc((size_t)BB * HH * NN * 4);
  __bf16* attn2 = (__bf16*)alloc((size_t)BB * HH * NN * NN * 2);
  if (off > ws_size) return;                   // workspace too small: bail out

  const long sNC = (long)NN * CC;              // per-batch stride in [B,N,C]
  const long sSS = (long)NN * NN;              // per-(b,h) stride of N x N maps
  const long sVt = (long)DD * NN;              // per-(b,h) stride of v_t

  // 1) LayerNorm
  ln_kernel<<<BB * NN, 256, 0, stream>>>(x, ln_g, ln_b, x1);
  // 2) weight conversion
  wcvt_kernel<<<(CC * CC + 255) / 256, 256, 0, stream>>>(proj_w, Wbf, CC * CC);
  // 3) fused depthwise convs
  dwconv_kernel<<<dim3(CC / 64, NN / 64, BB), 256, 0, stream>>>(
      x1, w0_1, b0_1, w0_2, b0_2, w1_1, b1_1, w1_2, b1_2, w2_1, b2_1, w2_2, b2_2,
      conv1, conv2);
  // 4) projections (bf16 WMMA, f32 out)
  gemm_tn_wmma<2, 4, false><<<dim3(CC / 128, (BB * NN) / 64, 1), 256, 0, stream>>>(
      conv1, Wbf, out1, BB * NN, CC, CC, CC, CC, CC,
      0, 0, 0, 0, 0, 0, 1, proj_b, 1.0f, nullptr, nullptr, 0);
  gemm_tn_wmma<2, 4, false><<<dim3(CC / 128, (BB * NN) / 64, 1), 256, 0, stream>>>(
      conv2, Wbf, out2, BB * NN, CC, CC, CC, CC, CC,
      0, 0, 0, 0, 0, 0, 1, proj_b, 1.0f, nullptr, nullptr, 0);
  // 5) per-head norms + transposed V
  normsplit_kernel<<<BB * NN, 256, 0, stream>>>(out1, out2, o1n, o2n, v1t, v2t);
  // 6) S = q1 @ k1^T per (b,h)  ->  d_out attn1 slot (pre-softmax)
  gemm_tn_wmma<2, 4, false><<<dim3(NN / 128, NN / 64, BB * HH), 256, 0, stream>>>(
      o2n, o1n, S, NN, NN, DD, CC, CC, NN,
      sNC, DD, sNC, DD, (long)HH * sSS, sSS, HH,
      nullptr, 0.f, nullptr, nullptr, 0);
  // 7) column softmax -> attn2 (bf16)
  colstat_kernel<<<dim3(NN / 256, BB * HH), 256, 0, stream>>>(S, cmax, csum);
  attn2_kernel<<<dim3(NN / 64, NN / 64, BB * HH), 256, 0, stream>>>(S, cmax, csum, attn2);
  // 8) row softmax in place -> attn1 final output
  rowsoftmax_kernel<<<BB * HH * NN, 256, 0, stream>>>(S);
  // 9) m = attn1 @ v1 + q1  (A is f32, converted to bf16 on the fly)
  gemm_tn_wmma<4, 2, true><<<dim3(DD / 64, NN / 128, BB * HH), 256, 0, stream>>>(
      S, v1t, mbuf, NN, DD, NN, NN, NN, CC,
      (long)HH * sSS, sSS, (long)HH * sVt, sVt, sNC, DD, HH,
      nullptr, 0.f, nullptr, o2n, 0);
  //    m += attn2 @ v2 + q2
  gemm_tn_wmma<4, 2, false><<<dim3(DD / 64, NN / 128, BB * HH), 256, 0, stream>>>(
      attn2, v2t, mbuf, NN, DD, NN, NN, NN, CC,
      (long)HH * sSS, sSS, (long)HH * sVt, sVt, sNC, DD, HH,
      nullptr, 0.f, nullptr, o1n, 1);
  // 10) out = m @ W^T + 2*proj_b + x
  gemm_tn_wmma<2, 4, true><<<dim3(CC / 128, (BB * NN) / 64, 1), 256, 0, stream>>>(
      mbuf, Wbf, out, BB * NN, CC, CC, CC, CC, CC,
      0, 0, 0, 0, 0, 0, 1, proj_b, 2.0f, x, nullptr, 0);
}
